// MoE_86139864088816
// MI455X (gfx1250) — matrix operand order repaired and checked by reading, not compiled, not run on hardware
//
#include <hip/hip_runtime.h>

// ---------------- problem constants ----------------
constexpr int kDim      = 2048;
constexpr int kInter    = 1408;
constexpr int kShInter  = 2816;
constexpr int kExperts  = 16;
constexpr int kTok      = 8192;          // 4 * 2048 tokens
constexpr int kCap      = 2 * kTok + kExperts * 64;  // 17408, multiple of 64

typedef __attribute__((ext_vector_type(16))) __bf16 v16bf;
typedef __attribute__((ext_vector_type(8)))  __bf16 v8bf;
typedef __attribute__((ext_vector_type(8)))  float  v8f;
typedef __attribute__((ext_vector_type(4)))  float  v4f;

// fp32 -> bf16, round-to-nearest-even (used in memory-bound cast/epilogue paths)
__device__ __forceinline__ unsigned short f2bf(float f) {
  unsigned int u = __builtin_bit_cast(unsigned int, f);
  u += 0x7fffu + ((u >> 16) & 1u);
  return (unsigned short)(u >> 16);
}

__device__ __forceinline__ v8f wmma_bf16(v16bf a, v16bf b, v8f c) {
  // D = A(16x32) * B(32x16) + C, fp32 accumulate
  return __builtin_amdgcn_wmma_f32_16x16x32_bf16(
      /*neg_a=*/false, a, /*neg_b=*/false, b,
      /*c_mod=*/(short)0, c, /*reuse_a=*/false, /*reuse_b=*/false);
}

// A fragment (16x32 bf16). Lane L serves row m = L&15.
// Lanes 0-15 hold K=[0,16), lanes 16-31 hold K=[16,32).
__device__ __forceinline__ v16bf load_a_frag(const unsigned short* rowp, int k0, int lane) {
  const int kh = (lane >> 4) << 4;
  v8bf lo = *reinterpret_cast<const v8bf*>(rowp + k0 + kh);
  v8bf hi = *reinterpret_cast<const v8bf*>(rowp + k0 + kh + 8);
  return __builtin_shufflevector(lo, hi, 0,1,2,3,4,5,6,7,8,9,10,11,12,13,14,15);
}

// B fragment (32x16) from pre-cast bf16 weights. Lane L serves column n = L&15;
// lanes 0-15 K=[k0,k0+16), lanes 16-31 K=[k0+16,k0+32): 16 contiguous elements.
__device__ __forceinline__ v16bf load_b_frag_bf16(const unsigned short* wrow, int k0, int lane) {
  const unsigned short* p = wrow + k0 + ((lane >> 4) << 4);
  v8bf lo = *reinterpret_cast<const v8bf*>(p);
  v8bf hi = *reinterpret_cast<const v8bf*>(p + 8);
  return __builtin_shufflevector(lo, hi, 0,1,2,3,4,5,6,7,8,9,10,11,12,13,14,15);
}

// Fallback: B fragment from fp32 weights, converted on the fly.
__device__ __forceinline__ v16bf load_b_frag_f32(const float* wrow, int k0, int lane) {
  const float* p = wrow + k0 + ((lane >> 4) << 4);
  const v4f* q = reinterpret_cast<const v4f*>(p);
  v4f f0 = q[0], f1 = q[1], f2 = q[2], f3 = q[3];
  v16bf r;
#pragma unroll
  for (int i = 0; i < 4; ++i) {
    r[i]      = __builtin_bit_cast(__bf16, f2bf(f0[i]));
    r[4 + i]  = __builtin_bit_cast(__bf16, f2bf(f1[i]));
    r[8 + i]  = __builtin_bit_cast(__bf16, f2bf(f2[i]));
    r[12 + i] = __builtin_bit_cast(__bf16, f2bf(f3[i]));
  }
  return r;
}

// ---------------- small prep kernels ----------------

__global__ void zero_kernel(float* __restrict__ out, long n, int* __restrict__ counts) {
  long i = (long)blockIdx.x * blockDim.x + threadIdx.x;
  if (i < n) out[i] = 0.0f;
  if (i < kExperts) counts[i] = 0;
}

__global__ void cast_kernel(const float* __restrict__ x, unsigned short* __restrict__ xb, int n) {
  int i = blockIdx.x * blockDim.x + threadIdx.x;
  if (i < n) xb[i] = f2bf(x[i]);
}

// One wave per token: 16 gate logits, softmax, top-2, append to expert lists.
__global__ __launch_bounds__(256)
void gate_kernel(const float* __restrict__ x, const float* __restrict__ gw,
                 int* __restrict__ counts, int* __restrict__ lists,
                 float* __restrict__ gwt) {
  const int lane = threadIdx.x & 31;
  const int t = blockIdx.x * 8 + (threadIdx.x >> 5);
  const float* xr = x + (size_t)t * kDim;

  float acc[kExperts];
#pragma unroll
  for (int e = 0; e < kExperts; ++e) acc[e] = 0.0f;
  for (int k = lane; k < kDim; k += 32) {
    float xv = xr[k];
#pragma unroll
    for (int e = 0; e < kExperts; ++e) acc[e] += xv * gw[e * kDim + k];
  }
#pragma unroll
  for (int e = 0; e < kExperts; ++e)
    for (int off = 16; off; off >>= 1) acc[e] += __shfl_xor(acc[e], off, 32);

  if (lane == 0) {
    float mx = acc[0];
#pragma unroll
    for (int e = 1; e < kExperts; ++e) mx = fmaxf(mx, acc[e]);
    float p[kExperts], sum = 0.0f;
#pragma unroll
    for (int e = 0; e < kExperts; ++e) { p[e] = __expf(acc[e] - mx); sum += p[e]; }
    float inv = 1.0f / sum;
    int i1 = 0; float v1 = p[0];
#pragma unroll
    for (int e = 1; e < kExperts; ++e) if (p[e] > v1) { v1 = p[e]; i1 = e; }
    int i2 = (i1 == 0) ? 1 : 0; float v2 = p[i2];
#pragma unroll
    for (int e = 0; e < kExperts; ++e)
      if (e != i1 && p[e] > v2) { v2 = p[e]; i2 = e; }

    int s1 = atomicAdd(&counts[i1], 1);
    lists[i1 * kTok + s1] = t; gwt[i1 * kTok + s1] = v1 * inv;
    int s2 = atomicAdd(&counts[i2], 1);
    lists[i2 * kTok + s2] = t; gwt[i2 * kTok + s2] = v2 * inv;
  }
}

// Per-expert slot offsets, padded to 64 so each 64-row GEMM tile is single-expert.
__global__ void offsets_kernel(const int* __restrict__ counts, int* __restrict__ offs) {
  if (blockIdx.x == 0 && threadIdx.x == 0) {
    int run = 0;
    for (int e = 0; e < kExperts; ++e) { offs[e] = run; run += (counts[e] + 63) & ~63; }
    offs[kExperts] = run;
  }
}

__global__ void compact_kernel(const int* __restrict__ offs, const int* __restrict__ counts,
                               const int* __restrict__ lists, const float* __restrict__ gwt,
                               int* __restrict__ stok, float* __restrict__ swt,
                               int* __restrict__ sexp) {
  int s = blockIdx.x * blockDim.x + threadIdx.x;
  if (s >= kCap) return;
  int e = 0; bool found = false;
#pragma unroll
  for (int i = 0; i < kExperts; ++i)
    if (s >= offs[i] && s < offs[i + 1]) { e = i; found = true; break; }
  int tok = 0; float w = 0.0f;
  if (found) {
    int local = s - offs[e];
    if (local < counts[e]) { tok = lists[e * kTok + local]; w = gwt[e * kTok + local]; }
  }
  stok[s] = tok; swt[s] = w; sexp[s] = e;
}

// ---------------- WMMA GEMM kernels ----------------
// Block: 256 threads = 8 waves; tile M=64 x N=128; wave tile 32x32 (2x2 WMMA frags).
// WB16 = true: weights already bf16 (no conversion VALU in the hot loop).

template <bool WB16>
__global__ __launch_bounds__(256)
void gemm_up_kernel(const unsigned short* __restrict__ A, const int* __restrict__ rowmap,
                    const int* __restrict__ slot_e,
                    const void* __restrict__ Wa, const void* __restrict__ Wb,
                    unsigned long long wstride, int K, int N,
                    unsigned short* __restrict__ H) {
  const int lane = threadIdx.x & 31;
  const int wave = threadIdx.x >> 5;
  const int waveM = blockIdx.y * 64 + (wave >> 2) * 32;
  const int waveN = blockIdx.x * 128 + (wave & 3) * 32;

  const int e = slot_e ? slot_e[blockIdx.y * 64] : 0;
  const size_t esz = WB16 ? 2 : 4;

  const unsigned short* arow[2];
#pragma unroll
  for (int mi = 0; mi < 2; ++mi) {
    int row = waveM + mi * 16 + (lane & 15);
    int tok = rowmap ? rowmap[row] : row;
    arow[mi] = A + (size_t)tok * K;
  }
  const char *browa[2], *browb[2];
#pragma unroll
  for (int ni = 0; ni < 2; ++ni) {
    size_t n = (size_t)(waveN + ni * 16 + (lane & 15));
    browa[ni] = (const char*)Wa + ((unsigned long long)e * wstride + n * K) * esz;
    browb[ni] = (const char*)Wb + ((unsigned long long)e * wstride + n * K) * esz;
  }

  const v8f vz = {0.f, 0.f, 0.f, 0.f, 0.f, 0.f, 0.f, 0.f};
  v8f acc1[2][2], acc3[2][2];
#pragma unroll
  for (int mi = 0; mi < 2; ++mi)
#pragma unroll
    for (int ni = 0; ni < 2; ++ni) { acc1[mi][ni] = vz; acc3[mi][ni] = vz; }

  for (int k0 = 0; k0 < K; k0 += 32) {
    v16bf af[2];
#pragma unroll
    for (int mi = 0; mi < 2; ++mi) af[mi] = load_a_frag(arow[mi], k0, lane);
#pragma unroll
    for (int ni = 0; ni < 2; ++ni) {
      v16bf b1, b3;
      if constexpr (WB16) {
        b1 = load_b_frag_bf16((const unsigned short*)browa[ni], k0, lane);
        b3 = load_b_frag_bf16((const unsigned short*)browb[ni], k0, lane);
      } else {
        b1 = load_b_frag_f32((const float*)browa[ni], k0, lane);
        b3 = load_b_frag_f32((const float*)browb[ni], k0, lane);
      }
#pragma unroll
      for (int mi = 0; mi < 2; ++mi) {
        acc1[mi][ni] = wmma_bf16(af[mi], b1, acc1[mi][ni]);
        acc3[mi][ni] = wmma_bf16(af[mi], b3, acc3[mi][ni]);
      }
    }
  }

  // fused SiLU(g1) * g3 -> bf16.  D layout: lane L, vgpr r -> (m = r + 8*(L>=16), n = L&15)
#pragma unroll
  for (int mi = 0; mi < 2; ++mi)
#pragma unroll
    for (int ni = 0; ni < 2; ++ni) {
      int n = waveN + ni * 16 + (lane & 15);
#pragma unroll
      for (int r = 0; r < 8; ++r) {
        int m = waveM + mi * 16 + ((lane >> 4) << 3) + r;
        float g = acc1[mi][ni][r];
        float h = (g / (1.0f + __expf(-g))) * acc3[mi][ni][r];
        H[(size_t)m * N + n] = f2bf(h);
      }
    }
}

// Down-projection: out[token[s]][n] += (H_s . W2[n]) * weight[s]   (N fixed = kDim)
template <bool WB16>
__global__ __launch_bounds__(256)
void gemm_down_kernel(const unsigned short* __restrict__ H, const int* __restrict__ rowmap,
                      const float* __restrict__ sw, const int* __restrict__ slot_e,
                      const void* __restrict__ W2, unsigned long long wstride, int K,
                      float* __restrict__ out) {
  const int lane = threadIdx.x & 31;
  const int wave = threadIdx.x >> 5;
  const int waveM = blockIdx.y * 64 + (wave >> 2) * 32;
  const int waveN = blockIdx.x * 128 + (wave & 3) * 32;

  const int e = slot_e ? slot_e[blockIdx.y * 64] : 0;
  const size_t esz = WB16 ? 2 : 4;

  const unsigned short* arow[2];
#pragma unroll
  for (int mi = 0; mi < 2; ++mi)
    arow[mi] = H + (size_t)(waveM + mi * 16 + (lane & 15)) * K;
  const char* brow[2];
#pragma unroll
  for (int ni = 0; ni < 2; ++ni) {
    size_t n = (size_t)(waveN + ni * 16 + (lane & 15));
    brow[ni] = (const char*)W2 + ((unsigned long long)e * wstride + n * K) * esz;
  }

  const v8f vz = {0.f, 0.f, 0.f, 0.f, 0.f, 0.f, 0.f, 0.f};
  v8f acc[2][2];
#pragma unroll
  for (int mi = 0; mi < 2; ++mi)
#pragma unroll
    for (int ni = 0; ni < 2; ++ni) acc[mi][ni] = vz;

  for (int k0 = 0; k0 < K; k0 += 32) {
    v16bf af[2];
#pragma unroll
    for (int mi = 0; mi < 2; ++mi) af[mi] = load_a_frag(arow[mi], k0, lane);
#pragma unroll
    for (int ni = 0; ni < 2; ++ni) {
      v16bf b;
      if constexpr (WB16) b = load_b_frag_bf16((const unsigned short*)brow[ni], k0, lane);
      else                b = load_b_frag_f32((const float*)brow[ni], k0, lane);
#pragma unroll
      for (int mi = 0; mi < 2; ++mi) acc[mi][ni] = wmma_bf16(af[mi], b, acc[mi][ni]);
    }
  }

#pragma unroll
  for (int mi = 0; mi < 2; ++mi)
#pragma unroll
    for (int ni = 0; ni < 2; ++ni) {
      int n = waveN + ni * 16 + (lane & 15);
#pragma unroll
      for (int r = 0; r < 8; ++r) {
        int m = waveM + mi * 16 + ((lane >> 4) << 3) + r;
        int tok = rowmap ? rowmap[m] : m;
        float wgt = sw ? sw[m] : 1.0f;
        atomicAdd(out + (size_t)tok * kDim + n, acc[mi][ni][r] * wgt);
      }
    }
}

// ---------------- host launch ----------------
extern "C" void kernel_launch(void* const* d_in, const int* in_sizes, int n_in,
                              void* d_out, int out_size, void* d_ws, size_t ws_size,
                              hipStream_t stream) {
  const float* x   = (const float*)d_in[0];
  const float* gw  = (const float*)d_in[1];
  const float* w1  = (const float*)d_in[2];
  const float* w2  = (const float*)d_in[3];
  const float* w3  = (const float*)d_in[4];
  const float* sw1 = (const float*)d_in[5];
  const float* sw2 = (const float*)d_in[6];
  const float* sw3 = (const float*)d_in[7];
  float* out = (float*)d_out;

  unsigned char* base = (unsigned char*)d_ws;
  size_t off = 0;
  auto take = [&](size_t bytes) -> void* {
    void* p = base + off;
    off += (bytes + 255) & ~(size_t)255;
    return p;
  };
  int*   counts = (int*)take(kExperts * sizeof(int));
  int*   offs   = (int*)take((kExperts + 1) * sizeof(int));
  int*   lists  = (int*)take((size_t)kExperts * kTok * sizeof(int));
  float* gwt    = (float*)take((size_t)kExperts * kTok * sizeof(float));
  int*   stok   = (int*)take((size_t)kCap * sizeof(int));
  float* swt    = (float*)take((size_t)kCap * sizeof(float));
  int*   sexp   = (int*)take((size_t)kCap * sizeof(int));
  unsigned short* xb   = (unsigned short*)take((size_t)kTok * kDim * 2);
  unsigned short* hexp = (unsigned short*)take((size_t)kCap * kInter * 2);
  unsigned short* hsh  = (unsigned short*)take((size_t)kTok * kShInter * 2);

  // optional pre-cast bf16 weight buffers (~312 MB); used only if ws is big enough
  const int nExpW  = kExperts * kInter * kDim;   // per expert tensor (w1/w3); w2 same count
  const int nShUpW = kShInter * kDim;            // sw1/sw3; sw2 same count
  unsigned short* w1b  = (unsigned short*)take((size_t)nExpW * 2);
  unsigned short* w3b  = (unsigned short*)take((size_t)nExpW * 2);
  unsigned short* w2b  = (unsigned short*)take((size_t)nExpW * 2);
  unsigned short* sw1b = (unsigned short*)take((size_t)nShUpW * 2);
  unsigned short* sw3b = (unsigned short*)take((size_t)nShUpW * 2);
  unsigned short* sw2b = (unsigned short*)take((size_t)nShUpW * 2);
  const bool wb16 = (off <= ws_size);
  (void)in_sizes; (void)n_in; (void)out_size;

  const long nOut = (long)kTok * kDim;
  zero_kernel<<<(unsigned)((nOut + 255) / 256), 256, 0, stream>>>(out, nOut, counts);
  cast_kernel<<<(unsigned)((kTok * kDim + 255) / 256), 256, 0, stream>>>(x, xb, kTok * kDim);
  if (wb16) {
    cast_kernel<<<(unsigned)((nExpW + 255) / 256), 256, 0, stream>>>(w1, w1b, nExpW);
    cast_kernel<<<(unsigned)((nExpW + 255) / 256), 256, 0, stream>>>(w3, w3b, nExpW);
    cast_kernel<<<(unsigned)((nExpW + 255) / 256), 256, 0, stream>>>(w2, w2b, nExpW);
    cast_kernel<<<(unsigned)((nShUpW + 255) / 256), 256, 0, stream>>>(sw1, sw1b, nShUpW);
    cast_kernel<<<(unsigned)((nShUpW + 255) / 256), 256, 0, stream>>>(sw3, sw3b, nShUpW);
    cast_kernel<<<(unsigned)((nShUpW + 255) / 256), 256, 0, stream>>>(sw2, sw2b, nShUpW);
  }
  gate_kernel<<<kTok / 8, 256, 0, stream>>>(x, gw, counts, lists, gwt);
  offsets_kernel<<<1, 32, 0, stream>>>(counts, offs);
  compact_kernel<<<(kCap + 255) / 256, 256, 0, stream>>>(offs, counts, lists, gwt, stok, swt, sexp);

  const unsigned long long sE = (unsigned long long)kInter * kDim;  // per-expert stride (elements)
  const dim3 gUpE(kInter / 128, kCap / 64),  gDnE(kDim / 128, kCap / 64);
  const dim3 gUpS(kShInter / 128, kTok / 64), gDnS(kDim / 128, kTok / 64);

  if (wb16) {
    gemm_up_kernel<true><<<gUpE, 256, 0, stream>>>(xb, stok, sexp, w1b, w3b, sE, kDim, kInter, hexp);
    gemm_down_kernel<true><<<gDnE, 256, 0, stream>>>(hexp, stok, swt, sexp, w2b, sE, kInter, out);
    gemm_up_kernel<true><<<gUpS, 256, 0, stream>>>(xb, nullptr, nullptr, sw1b, sw3b, 0ull, kDim, kShInter, hsh);
    gemm_down_kernel<true><<<gDnS, 256, 0, stream>>>(hsh, nullptr, nullptr, nullptr, sw2b, 0ull, kShInter, out);
  } else {
    gemm_up_kernel<false><<<gUpE, 256, 0, stream>>>(xb, stok, sexp, w1, w3, sE, kDim, kInter, hexp);
    gemm_down_kernel<false><<<gDnE, 256, 0, stream>>>(hexp, stok, swt, sexp, w2, sE, kInter, out);
    gemm_up_kernel<false><<<gUpS, 256, 0, stream>>>(xb, nullptr, nullptr, sw1, sw3, 0ull, kDim, kShInter, hsh);
    gemm_down_kernel<false><<<gDnS, 256, 0, stream>>>(hsh, nullptr, nullptr, nullptr, sw2, 0ull, kShInter, out);
  }
}
